// VRNN_58385785422310
// MI455X (gfx1250) — matrix operand order repaired and checked by reading, not compiled, not run on hardware
//
#include <hip/hip_runtime.h>
#include <hip/hip_bf16.h>
#include <math.h>

// Problem constants (from reference)
#define T_STEPS 128
#define B_SZ    256
#define XDIM    88
#define HDIM    512
#define ZDIM    64
#define PN      16
#define PB      (PN * B_SZ)   // 4096 particle-batch rows

typedef __attribute__((ext_vector_type(16))) _Float16 v16h;
typedef __attribute__((ext_vector_type(8)))  _Float16 v8h;
typedef __attribute__((ext_vector_type(8)))  float    v8f;

__device__ __forceinline__ v8f vrnn_wmma(v16h a, v16h b, v8f c) {
  // D = A(16x32 f16) * B(32x16 f16) + C(16x16 f32)
  return __builtin_amdgcn_wmma_f32_16x16x32_f16(false, a, false, b, (short)0, c,
                                                false, false);
}

__device__ __forceinline__ float vrnn_sigmoid(float x) {
  return 1.0f / (1.0f + expf(-x));
}
__device__ __forceinline__ float vrnn_softplus(float x) {
  return (x > 20.0f) ? x : log1pf(expf(x));
}

// ---------------------------------------------------------------------------
// GEMM: O = act( A[M,K](f16) @ W[Npad,K](f16)^T + bias )
// block = 256 threads = 8 waves; block tile 128(M) x 256(N);
// wave tile 64(M) x 64(N) = 4x4 WMMA 16x16x32 tiles -> 16 WMMA : 16 b128 loads
// per K-step. A and B fragments are kept in independent register arrays so the
// whole load clause for a K-step can be in flight while WMMA groups retire
// under partial s_wait_loadcnt thresholds.
// Requires: M % 128 == 0, Npad % 256 == 0, K % 32 == 0 (buffers pre-padded).
// ---------------------------------------------------------------------------
__global__ __launch_bounds__(256, 1)
void vrnn_gemm_f16(const _Float16* __restrict__ A, int lda,
                   const _Float16* __restrict__ W, int K,
                   const float* __restrict__ bias,
                   float* __restrict__ O32, int ldo32,
                   _Float16* __restrict__ O16, int ldo16,
                   int Nreal, int act) {
  const int lane  = threadIdx.x & 31;
  const int wid   = threadIdx.x >> 5;
  const int waveM = wid & 1;   // 2 waves along M
  const int waveN = wid >> 1;  // 4 waves along N
  const int m0 = blockIdx.x * 128 + waveM * 64;
  const int n0 = blockIdx.y * 256 + waveN * 64;

  const int lhalf = lane >> 4;  // 0 or 1 (lane group)
  const int l16   = lane & 15;

  v8f acc[4][4] = {};

  for (int kk = 0; kk < K; kk += 32) {
    // A fragments: 16-bit A 16x32 layout — lane l holds row (.. + l%16);
    // halves e -> K = kk + lhalf*16 + e
    v16h afrag[4];
#pragma unroll
    for (int mt = 0; mt < 4; ++mt) {
      const _Float16* ap =
          A + (size_t)(m0 + mt * 16 + l16) * lda + kk + lhalf * 16;
      afrag[mt] = *(const v16h*)(ap);
    }
    // B fragments: lane l holds column N = n0+nt*16+l%16,
    // halves e -> K = kk + lhalf*16 + e  (one contiguous 32B chunk of W row).
    // All four loaded into independent registers up front.
    v16h bfrag[4];
#pragma unroll
    for (int nt = 0; nt < 4; ++nt) {
      const _Float16* bp =
          W + (size_t)(n0 + nt * 16 + l16) * K + kk + lhalf * 16;
      bfrag[nt] = *(const v16h*)bp;
    }
#pragma unroll
    for (int nt = 0; nt < 4; ++nt)
#pragma unroll
      for (int mt = 0; mt < 4; ++mt)
        acc[mt][nt] = vrnn_wmma(afrag[mt], bfrag[nt], acc[mt][nt]);
  }

  // Epilogue. C/D layout: VGPR v -> row = base + v + lhalf*8, col = base + l%16
#pragma unroll
  for (int nt = 0; nt < 4; ++nt) {
    const int col = n0 + nt * 16 + l16;
    const bool cok = (col < Nreal);
    const float bv = (bias && cok) ? bias[col] : 0.0f;
#pragma unroll
    for (int mt = 0; mt < 4; ++mt) {
#pragma unroll
      for (int v = 0; v < 8; ++v) {
        const int row = m0 + mt * 16 + lhalf * 8 + v;
        float val = acc[mt][nt][v] + bv;
        if (cok) {
          if (act == 1) val = fmaxf(val, 0.0f);
          if (O32) O32[(size_t)row * ldo32 + col] = val;
          if (O16) O16[(size_t)row * ldo16 + col] = (_Float16)val;
        }
      }
    }
  }
}

// ---------------------------------------------------------------------------
// Weight packing: dst[(rowOff+n)*Kdst + colOff + k] = src[n*Ksrc + k]
// ---------------------------------------------------------------------------
__global__ void vrnn_pack_w(const float* __restrict__ src,
                            _Float16* __restrict__ dst, int N, int Ksrc,
                            int Kdst, int rowOff, int colOff) {
  int idx = blockIdx.x * 256 + threadIdx.x;
  if (idx >= N * Ksrc) return;
  int n = idx / Ksrc, k = idx % Ksrc;
  dst[(size_t)(rowOff + n) * Kdst + colOff + k] = (_Float16)src[idx];
}

__global__ void vrnn_pack_b(const float* __restrict__ src,
                            float* __restrict__ dst, int n, int off) {
  int i = blockIdx.x * 256 + threadIdx.x;
  if (i < n) dst[off + i] = src[i];
}

__global__ void vrnn_conv_x(const float* __restrict__ xt,
                            _Float16* __restrict__ dst) {
  int idx = blockIdx.x * 256 + threadIdx.x;  // 256 x 96
  if (idx >= B_SZ * 96) return;
  int b = idx / 96, k = idx % 96;
  dst[idx] = (_Float16)((k < XDIM) ? xt[b * XDIM + k] : 0.0f);
}

__global__ void vrnn_fill_f32(float* p, float v, int n) {
  int i = blockIdx.x * 256 + threadIdx.x;
  if (i < n) p[i] = v;
}
__global__ void vrnn_fill_f16(_Float16* p, int n) {
  int i = blockIdx.x * 256 + threadIdx.x;
  if (i < n) p[i] = (_Float16)0.0f;
}

__global__ void vrnn_skip_flags(const float* __restrict__ x,
                                int* __restrict__ skip) {
  __shared__ int any;
  if (threadIdx.x == 0) any = 0;
  __syncthreads();
  const float* xt = x + (size_t)blockIdx.x * B_SZ * XDIM;
  int found = 0;
  for (int i = threadIdx.x; i < B_SZ * XDIM; i += 256)
    if (xt[i] != 0.0f) { found = 1; break; }
  if (found) any = 1;
  __syncthreads();
  if (threadIdx.x == 0) skip[blockIdx.x] = (any == 0) ? 1 : 0;
}

// dst[PB, ldd] cols [0,512) = s0[r%mod0], cols [512,1024) = s1[r%mod1]
__global__ void vrnn_concat2(_Float16* __restrict__ dst, int ldd,
                             const _Float16* __restrict__ s0, int mod0,
                             const _Float16* __restrict__ s1, int mod1) {
  int idx = blockIdx.x * 256 + threadIdx.x;
  if (idx >= PB * 1024) return;
  int r = idx >> 10, k = idx & 1023;
  _Float16 v = (k < HDIM) ? s0[(r % mod0) * HDIM + k]
                          : s1[(r % mod1) * HDIM + (k - HDIM)];
  dst[(size_t)r * ldd + k] = v;
}

// dst[PB,1536] = [ s0[r%mod0] | s1[r] | s2[r] ]   (LSTM input)
__global__ void vrnn_concat3(_Float16* __restrict__ dst,
                             const _Float16* __restrict__ s0, int mod0,
                             const _Float16* __restrict__ s1,
                             const _Float16* __restrict__ s2) {
  int idx = blockIdx.x * 256 + threadIdx.x;
  if (idx >= PB * 1536) return;
  int r = idx / 1536, k = idx % 1536;
  _Float16 v;
  if (k < HDIM)            v = s0[(r % mod0) * HDIM + k];
  else if (k < 2 * HDIM)   v = s1[r * HDIM + (k - HDIM)];
  else                     v = s2[r * HDIM + (k - 2 * HDIM)];
  dst[(size_t)r * 1536 + k] = v;
}

// Heads live in fused buffers: ems[r*128 + k] = enc_mu, [.. + 64 + k] = enc_s
// std = softplus(raw)+c (in place), z = mu + std*eps, write z32 + z16
__global__ void vrnn_z_prep(float* __restrict__ ems, float* __restrict__ pms,
                            const float* __restrict__ eps_t,
                            float* __restrict__ z32,
                            _Float16* __restrict__ z16) {
  int idx = blockIdx.x * 256 + threadIdx.x;  // PB*ZD
  if (idx >= PB * ZDIM) return;
  int r = idx / ZDIM, k = idx % ZDIM;
  float es = vrnn_softplus(ems[r * 128 + 64 + k]) + 1e-7f;
  float ps = vrnn_softplus(pms[r * 128 + 64 + k]) + 0.01f;
  ems[r * 128 + 64 + k] = es;
  pms[r * 128 + 64 + k] = ps;
  float z = ems[r * 128 + k] + es * eps_t[idx];
  z32[idx] = z;
  z16[idx] = (_Float16)z;
}

// la[r] = logN(z;pri) - logN(z;enc) + sum_j x*log(m)+(1-x)*log1p(-m)
__global__ void vrnn_row_logprobs(const float* __restrict__ z32,
                                  const float* __restrict__ ems,
                                  const float* __restrict__ pms,
                                  const float* __restrict__ dec_head,
                                  const float* __restrict__ x_t,
                                  float* __restrict__ la) {
  int r = blockIdx.x * 256 + threadIdx.x;
  if (r >= PB) return;
  float se = 0.0f, sp = 0.0f;
  for (int k = 0; k < ZDIM; ++k) {
    float z = z32[r * ZDIM + k];
    float emu = ems[r * 128 + k], estd = ems[r * 128 + 64 + k];
    float pmu = pms[r * 128 + k], pstd = pms[r * 128 + 64 + k];
    float de = (z - emu) / estd;
    float dp = (z - pmu) / pstd;
    se += de * de + 2.0f * logf(estd);
    sp += dp * dp + 2.0f * logf(pstd);
  }
  float la_g = -0.5f * (sp - se);  // 0.5*ZD*log(2pi) terms cancel
  int b = r & (B_SZ - 1);
  float dlp = 0.0f;
  for (int j = 0; j < XDIM; ++j) {
    float m = vrnn_sigmoid(dec_head[(size_t)r * 128 + j]);
    float xv = x_t[b * XDIM + j];
    dlp += xv * logf(m) + (1.0f - xv) * log1pf(-m);
  }
  la[r] = la_g + dlp;
}

// Sequential particle-weight recursion (faithful in-loop row updates)
__global__ void vrnn_particle(const float* __restrict__ la,
                              float* __restrict__ lw, float* __restrict__ acc,
                              const float* __restrict__ mask_t,
                              const int* __restrict__ skip_t) {
  int b = threadIdx.x;
  if (b >= B_SZ) return;
  if (*skip_t) return;
  float lw2[PN], lai[PN], hat[PN];
#pragma unroll
  for (int i = 0; i < PN; ++i) {
    lw2[i] = lw[i * B_SZ + b];
    lai[i] = la[i * B_SZ + b];
    hat[i] = 0.0f;
  }
#pragma unroll
  for (int i = 0; i < PN; ++i) {
#pragma unroll
    for (int j = 0; j < PN; ++j) hat[j] += expf(lw2[j] + lai[i]);
    lw2[i] += lai[i];
  }
  float m = mask_t[b];
#pragma unroll
  for (int j = 0; j < PN; ++j) {
    float lh = logf(hat[j]);
    acc[j * B_SZ + b] += lh * m;
    lw[j * B_SZ + b] = lw2[j] - lh;
  }
}

__global__ void vrnn_lstm(const float* __restrict__ gates,
                          float* __restrict__ h, float* __restrict__ c,
                          _Float16* __restrict__ h16,
                          const int* __restrict__ skip_t) {
  if (*skip_t) return;
  int idx = blockIdx.x * 256 + threadIdx.x;  // PB*HD
  if (idx >= PB * HDIM) return;
  int r = idx / HDIM, k = idx % HDIM;
  const float* g = gates + (size_t)r * 4 * HDIM;
  float gi = g[k], gf = g[HDIM + k], gg = g[2 * HDIM + k], go = g[3 * HDIM + k];
  float cn = vrnn_sigmoid(gf) * c[idx] + vrnn_sigmoid(gi) * tanhf(gg);
  float hn = vrnn_sigmoid(go) * tanhf(cn);
  c[idx] = cn;
  h[idx] = hn;
  h16[idx] = (_Float16)hn;
}

__global__ void vrnn_final(const float* __restrict__ acc,
                           float* __restrict__ out) {
  __shared__ float s[256];
  float t = 0.0f;
  for (int i = threadIdx.x; i < PB; i += 256) t += acc[i];
  s[threadIdx.x] = t;
  __syncthreads();
  for (int o = 128; o > 0; o >>= 1) {
    if (threadIdx.x < (unsigned)o) s[threadIdx.x] += s[threadIdx.x + o];
    __syncthreads();
  }
  if (threadIdx.x == 0) out[0] = -s[0];
}

// ---------------------------------------------------------------------------
// Host driver
// ---------------------------------------------------------------------------
extern "C" void kernel_launch(void* const* d_in, const int* in_sizes, int n_in,
                              void* d_out, int out_size, void* d_ws,
                              size_t ws_size, hipStream_t stream) {
  (void)in_sizes; (void)n_in; (void)out_size; (void)ws_size;
  const float* x    = (const float*)d_in[0];
  const float* mask = (const float*)d_in[1];
  const float* eps  = (const float*)d_in[2];
  const float* pxw1 = (const float*)d_in[3];  const float* pxb1 = (const float*)d_in[4];
  const float* pxw2 = (const float*)d_in[5];  const float* pxb2 = (const float*)d_in[6];
  const float* pzw  = (const float*)d_in[7];  const float* pzb  = (const float*)d_in[8];
  const float* encw1= (const float*)d_in[9];  const float* encb1= (const float*)d_in[10];
  const float* encw2= (const float*)d_in[11]; const float* encb2= (const float*)d_in[12];
  const float* emw  = (const float*)d_in[13]; const float* emb  = (const float*)d_in[14];
  const float* esw  = (const float*)d_in[15]; const float* esb  = (const float*)d_in[16];
  const float* prw  = (const float*)d_in[17]; const float* prb  = (const float*)d_in[18];
  const float* pmw  = (const float*)d_in[19]; const float* pmb  = (const float*)d_in[20];
  const float* psw  = (const float*)d_in[21]; const float* psb  = (const float*)d_in[22];
  const float* dw1  = (const float*)d_in[23]; const float* db1  = (const float*)d_in[24];
  const float* dw2  = (const float*)d_in[25]; const float* db2  = (const float*)d_in[26];
  const float* dmw  = (const float*)d_in[27]; const float* dmb  = (const float*)d_in[28];
  const float* wih  = (const float*)d_in[29]; const float* whh  = (const float*)d_in[30];

  char* ws = (char*)d_ws;
  size_t off = 0;
  auto alloc = [&](size_t bytes) -> char* {
    char* p = ws + off;
    off = (off + bytes + 255) & ~(size_t)255;
    return p;
  };

  // f16 weight buffers (N padded to mult of 256, K to mult of 32)
  _Float16* w_pxw1 = (_Float16*)alloc((size_t)512 * 96 * 2);
  _Float16* w_pxw2 = (_Float16*)alloc((size_t)512 * 512 * 2);
  _Float16* w_pzw  = (_Float16*)alloc((size_t)512 * 64 * 2);
  _Float16* w_enc1 = (_Float16*)alloc((size_t)512 * 1024 * 2);
  _Float16* w_enc2 = (_Float16*)alloc((size_t)512 * 512 * 2);
  _Float16* w_ems  = (_Float16*)alloc((size_t)256 * 512 * 2);  // [em; es]
  _Float16* w_pr   = (_Float16*)alloc((size_t)512 * 512 * 2);
  _Float16* w_pms  = (_Float16*)alloc((size_t)256 * 512 * 2);  // [pm; ps]
  _Float16* w_d1   = (_Float16*)alloc((size_t)512 * 1024 * 2);
  _Float16* w_d2   = (_Float16*)alloc((size_t)512 * 512 * 2);
  _Float16* w_dm   = (_Float16*)alloc((size_t)256 * 512 * 2);
  _Float16* w_ihh  = (_Float16*)alloc((size_t)2048 * 1536 * 2); // [wih | whh]
  float*    b_ems  = (float*)   alloc((size_t)128 * 4);
  float*    b_pms  = (float*)   alloc((size_t)128 * 4);

  // activations / state
  _Float16* x16    = (_Float16*)alloc((size_t)B_SZ * 96 * 2);
  _Float16* px_t16 = (_Float16*)alloc((size_t)B_SZ * HDIM * 2);
  _Float16* phx16  = (_Float16*)alloc((size_t)B_SZ * HDIM * 2);
  float*    h32    = (float*)   alloc((size_t)PB * HDIM * 4);
  float*    c32    = (float*)   alloc((size_t)PB * HDIM * 4);
  _Float16* h16    = (_Float16*)alloc((size_t)PB * HDIM * 2);
  _Float16* cat16  = (_Float16*)alloc((size_t)PB * 1024 * 2);
  _Float16* cat16b = (_Float16*)alloc((size_t)PB * 1536 * 2);
  _Float16* t16a   = (_Float16*)alloc((size_t)PB * HDIM * 2);
  _Float16* t16b   = (_Float16*)alloc((size_t)PB * HDIM * 2);
  float*    ems32  = (float*)   alloc((size_t)PB * 128 * 4);
  float*    pms32  = (float*)   alloc((size_t)PB * 128 * 4);
  float*    z32    = (float*)   alloc((size_t)PB * ZDIM * 4);
  _Float16* z16    = (_Float16*)alloc((size_t)PB * ZDIM * 2);
  _Float16* phiz16 = (_Float16*)alloc((size_t)PB * HDIM * 2);
  float*    dechd  = (float*)   alloc((size_t)PB * 128 * 4);
  float*    gates  = (float*)   alloc((size_t)PB * 4 * HDIM * 4);
  float*    la     = (float*)   alloc((size_t)PB * 4);
  float*    lw     = (float*)   alloc((size_t)PB * 4);
  float*    accb   = (float*)   alloc((size_t)PB * 4);
  int*      skip   = (int*)     alloc((size_t)T_STEPS * 4);

  auto zfill16 = [&](_Float16* p, size_t n) {
    vrnn_fill_f16<<<(int)((n + 255) / 256), 256, 0, stream>>>(p, (int)n);
  };
  auto pack = [&](const float* src, _Float16* dst, int N, int Ks, int Kdst,
                  int rowOff, int colOff) {
    int cnt = N * Ks;
    vrnn_pack_w<<<(cnt + 255) / 256, 256, 0, stream>>>(src, dst, N, Ks, Kdst,
                                                       rowOff, colOff);
  };
  auto gemm = [&](const _Float16* A, int lda, const _Float16* W, int K,
                  const float* bias, float* O32, int ldo32, _Float16* O16,
                  int ldo16, int M, int Npad, int Nreal, int act) {
    dim3 grid(M / 128, Npad / 256);
    vrnn_gemm_f16<<<grid, 256, 0, stream>>>(A, lda, W, K, bias, O32, ldo32,
                                            O16, ldo16, Nreal, act);
  };

  // ---- once per launch: weight packing, skip flags, state init ----
  zfill16(w_pxw1, (size_t)512 * 96);
  pack(pxw1, w_pxw1, 512, 88, 96, 0, 0);
  pack(pxw2, w_pxw2, 512, 512, 512, 0, 0);
  pack(pzw,  w_pzw,  512, 64, 64, 0, 0);
  pack(encw1,w_enc1, 512, 1024, 1024, 0, 0);
  pack(encw2,w_enc2, 512, 512, 512, 0, 0);
  zfill16(w_ems, (size_t)256 * 512);
  pack(emw, w_ems, 64, 512, 512, 0, 0);
  pack(esw, w_ems, 64, 512, 512, 64, 0);
  pack(prw, w_pr, 512, 512, 512, 0, 0);
  zfill16(w_pms, (size_t)256 * 512);
  pack(pmw, w_pms, 64, 512, 512, 0, 0);
  pack(psw, w_pms, 64, 512, 512, 64, 0);
  pack(dw1, w_d1, 512, 1024, 1024, 0, 0);
  pack(dw2, w_d2, 512, 512, 512, 0, 0);
  zfill16(w_dm, (size_t)256 * 512);
  pack(dmw, w_dm, 88, 512, 512, 0, 0);
  pack(wih, w_ihh, 2048, 1024, 1536, 0, 0);     // cols [0,1024)
  pack(whh, w_ihh, 2048, 512, 1536, 0, 1024);   // cols [1024,1536)
  vrnn_pack_b<<<1, 256, 0, stream>>>(emb, b_ems, 64, 0);
  vrnn_pack_b<<<1, 256, 0, stream>>>(esb, b_ems, 64, 64);
  vrnn_pack_b<<<1, 256, 0, stream>>>(pmb, b_pms, 64, 0);
  vrnn_pack_b<<<1, 256, 0, stream>>>(psb, b_pms, 64, 64);

  vrnn_skip_flags<<<T_STEPS, 256, 0, stream>>>(x, skip);

  int nh = PB * HDIM;
  vrnn_fill_f32<<<(nh + 255) / 256, 256, 0, stream>>>(h32, 0.0f, nh);
  vrnn_fill_f32<<<(nh + 255) / 256, 256, 0, stream>>>(c32, 0.0f, nh);
  zfill16(h16, (size_t)nh);
  vrnn_fill_f32<<<(PB + 255) / 256, 256, 0, stream>>>(lw, -logf((float)PN), PB);
  vrnn_fill_f32<<<(PB + 255) / 256, 256, 0, stream>>>(accb, 0.0f, PB);

  // ---- sequential scan over timesteps ----
  for (int t = 0; t < T_STEPS; ++t) {
    const float* x_t    = x + (size_t)t * B_SZ * XDIM;
    const float* mask_t = mask + (size_t)t * B_SZ;
    const float* eps_t  = eps + (size_t)t * PN * B_SZ * ZDIM;
    const int*   skip_t = skip + t;

    // phi_x: [B,88]->512->512 (relu)
    vrnn_conv_x<<<(B_SZ * 96 + 255) / 256, 256, 0, stream>>>(x_t, x16);
    gemm(x16, 96, w_pxw1, 96, pxb1, nullptr, 0, px_t16, HDIM, B_SZ, 512, 512, 1);
    gemm(px_t16, HDIM, w_pxw2, 512, pxb2, nullptr, 0, phx16, HDIM, B_SZ, 512, 512, 1);

    // enc: [phx_bcast | h] -> 512 -> 512 (relu) -> fused [mu;std_raw] head
    vrnn_concat2<<<(PB * 1024 + 255) / 256, 256, 0, stream>>>(cat16, 1024, phx16, B_SZ, h16, PB);
    gemm(cat16, 1024, w_enc1, 1024, encb1, nullptr, 0, t16a, HDIM, PB, 512, 512, 1);
    gemm(t16a, HDIM, w_enc2, 512, encb2, nullptr, 0, t16b, HDIM, PB, 512, 512, 1);
    gemm(t16b, HDIM, w_ems, 512, b_ems, ems32, 128, nullptr, 0, PB, 256, 128, 0);

    // prior: h -> 512 (relu) -> fused [mu;std_raw] head
    gemm(h16, HDIM, w_pr, 512, prb, nullptr, 0, t16a, HDIM, PB, 512, 512, 1);
    gemm(t16a, HDIM, w_pms, 512, b_pms, pms32, 128, nullptr, 0, PB, 256, 128, 0);

    // z = mu + softplus(s)*eps ; phi_z: [PB,64]->512 (relu)
    vrnn_z_prep<<<(PB * ZDIM + 255) / 256, 256, 0, stream>>>(ems32, pms32,
                                                             eps_t, z32, z16);
    gemm(z16, ZDIM, w_pzw, 64, pzb, nullptr, 0, phiz16, HDIM, PB, 512, 512, 1);

    // dec: [phi_z | h] -> 512 -> 512 (relu) -> 88 head (raw; sigmoid later)
    vrnn_concat2<<<(PB * 1024 + 255) / 256, 256, 0, stream>>>(cat16, 1024, phiz16, PB, h16, PB);
    gemm(cat16, 1024, w_d1, 1024, db1, nullptr, 0, t16a, HDIM, PB, 512, 512, 1);
    gemm(t16a, HDIM, w_d2, 512, db2, nullptr, 0, t16b, HDIM, PB, 512, 512, 1);
    gemm(t16b, HDIM, w_dm, 512, dmb, dechd, 128, nullptr, 0, PB, 256, XDIM, 0);

    // log-probs + particle weight recursion
    vrnn_row_logprobs<<<(PB + 255) / 256, 256, 0, stream>>>(z32, ems32, pms32,
                                                            dechd, x_t, la);
    vrnn_particle<<<1, 256, 0, stream>>>(la, lw, accb, mask_t, skip_t);

    // LSTM fused: gates = [phx | phi_z | h] @ [wih | whh]^T  (K=1536, one GEMM)
    vrnn_concat3<<<(PB * 1536 + 255) / 256, 256, 0, stream>>>(cat16b, phx16,
                                                              B_SZ, phiz16, h16);
    gemm(cat16b, 1536, w_ihh, 1536, nullptr, gates, 4 * HDIM, nullptr, 0, PB,
         2048, 2048, 0);
    vrnn_lstm<<<(PB * HDIM + 255) / 256, 256, 0, stream>>>(gates, h32, c32,
                                                           h16, skip_t);
  }

  vrnn_final<<<1, 256, 0, stream>>>(accb, (float*)d_out);
}
